// DAttentionBaseline_1726576854812
// MI455X (gfx1250) — matrix operand order repaired and checked by reading, not compiled, not run on hardware
//
#include <hip/hip_runtime.h>
#include <hip/hip_bf16.h>

// ---------------------------------------------------------------------------
// Deformable attention (DAT) forward for MI455X / gfx1250, wave32 + WMMA.
// All matmuls use v_wmma_f32_16x16x32_f16 (f16 operands, f32 accumulate).
// Operands are pre-converted to f16 staging buffers once, so inner loops are
// pure b128 loads + WMMA (no per-iteration v_cvt, half the memory traffic).
// ---------------------------------------------------------------------------

typedef __attribute__((ext_vector_type(16))) _Float16 v16h;
typedef __attribute__((ext_vector_type(8)))  float    v8f;

#define NCH   512     // NC
#define HWPIX 1024    // H*W
#define NS    256     // Hs*Ws sampled points
#define HH    32
#define WW    32
#define DHD   64
#define NHEAD 8
#define NGRP  4
#define CGRP  128
#define TABW  63      // 2H-1

__device__ __forceinline__ v8f wmma_f16(v16h a, v16h b, v8f c) {
  // 8 args: (neg_a, A, neg_b, B, c_mod, C, reuse_a, reuse_b)
  return __builtin_amdgcn_wmma_f32_16x16x32_f16(false, a, false, b, (short)0, c, false, false);
}

// A-matrix (16x32 f16) per-lane packing: lanes 0-15 hold K = {0..7, 16..23},
// lanes 16-31 hold K = {8..15, 24..31}  ->  two 8-half runs at 8*hi, 16+8*hi.
__device__ __forceinline__ v16h ldA(const _Float16* p, int hi) {
  union { uint4 u[2]; v16h v; } t;
  t.u[0] = *(const uint4*)(p + 8 * hi);
  t.u[1] = *(const uint4*)(p + 16 + 8 * hi);
  return t.v;
}
// B-matrix (32x16 f16) per-lane packing: lanes 0-15 hold K = {0..7, 16..23},
// lanes 16-31 hold K = {8..15, 24..31}  ->  two 8-half runs at 8*hi, 16+8*hi.
__device__ __forceinline__ v16h ldB(const _Float16* p, int hi) {
  union { uint4 u[2]; v16h v; } t;
  t.u[0] = *(const uint4*)(p + 8 * hi);
  t.u[1] = *(const uint4*)(p + 16 + 8 * hi);
  return t.v;
}

__device__ __forceinline__ float hmax16(float v) {
#pragma unroll
  for (int m = 1; m < 16; m <<= 1) v = fmaxf(v, __shfl_xor(v, m, 32));
  return v;
}
__device__ __forceinline__ float hsum16(float v) {
#pragma unroll
  for (int m = 1; m < 16; m <<= 1) v += __shfl_xor(v, m, 32);
  return v;
}

// ---------------------------------------------------------------------------
// One-shot f32 -> f16 conversion (weights).
// ---------------------------------------------------------------------------
__global__ __launch_bounds__(256)
void k_cvt_f16(const float* __restrict__ src, _Float16* __restrict__ dst, int n) {
  int i = blockIdx.x * 256 + threadIdx.x;
  if (i < n) dst[i] = (_Float16)src[i];
}

// x[b][c][p] (f32, channel-major) -> x_t[b][p][c] (f16, pixel-major)
__global__ __launch_bounds__(256)
void k_xpose(const float* __restrict__ x, _Float16* __restrict__ x_t) {
  int c = blockIdx.x * 256 + threadIdx.x;
  int p = blockIdx.y, b = blockIdx.z;
  x_t[((size_t)b * HWPIX + p) * NCH + c] =
      (_Float16)x[((size_t)b * NCH + c) * HWPIX + p];
}

// ---------------------------------------------------------------------------
// q = x . Wq^T + bq.  A = x_t[b][p][c] (f16), B = Wq (f16).
// Writes q_t (f32, for the LN-sensitive offset net) and q_h (f16, for attn).
// ---------------------------------------------------------------------------
__global__ __launch_bounds__(32)
void k_gemm_q(const _Float16* __restrict__ x_t, const _Float16* __restrict__ wq,
              const float* __restrict__ bq, float* __restrict__ q_t,
              _Float16* __restrict__ q_h) {
  int lane = threadIdx.x, col = lane & 15, hi = lane >> 4;
  int nb = blockIdx.x * 16;   // out-channel tile (N)
  int pb = blockIdx.y * 16;   // pixel tile (M)
  int b  = blockIdx.z;
  const _Float16* arow = x_t + ((size_t)b * HWPIX + pb + col) * NCH;
  const _Float16* brow = wq + (size_t)(nb + col) * NCH;
  v8f acc = {};
  for (int kb = 0; kb < NCH; kb += 32)
    acc = wmma_f16(ldA(arow + kb, hi), ldB(brow + kb, hi), acc);
  float bias = bq[nb + col];
  float*    d32 = q_t + (size_t)b * HWPIX * NCH;
  _Float16* d16 = q_h + (size_t)b * HWPIX * NCH;
#pragma unroll
  for (int r = 0; r < 8; ++r) {
    float v = acc[r] + bias;
    size_t idx = (size_t)(pb + r + 8 * hi) * NCH + nb + col;
    d32[idx] = v;
    d16[idx] = (_Float16)v;
  }
}

// ---------------------------------------------------------------------------
// Offset network: depthwise 3x3 stride-2 conv -> LN(channels) -> GELU -> 1x1
// -> tanh * 2/15 -> +ref.  Writes pos (ws + d_out) and ref (d_out).
// ---------------------------------------------------------------------------
__global__ __launch_bounds__(256)
void k_offset(const float* __restrict__ q_t,
              const float* __restrict__ dw_w, const float* __restrict__ dw_b,
              const float* __restrict__ ln_g, const float* __restrict__ ln_b,
              const float* __restrict__ pw_w,
              float* __restrict__ posw, float* __restrict__ out_pos,
              float* __restrict__ out_ref) {
  int b = blockIdx.x >> 2, g = blockIdx.x & 3;
  int t = threadIdx.x;
  int hs = t >> 4, ws = t & 15;
  const float* qb = q_t + (size_t)b * HWPIX * NCH + g * CGRP;

  auto conv = [&](int c) -> float {
    float v = dw_b[c];
#pragma unroll
    for (int kh = 0; kh < 3; ++kh) {
      int ih = 2 * hs - 1 + kh;
      if (ih < 0 || ih >= HH) continue;
#pragma unroll
      for (int kw = 0; kw < 3; ++kw) {
        int iw = 2 * ws - 1 + kw;
        if (iw < 0 || iw >= WW) continue;
        v += dw_w[c * 9 + kh * 3 + kw] * qb[(size_t)(ih * WW + iw) * NCH + c];
      }
    }
    return v;
  };

  float sum = 0.f, sq = 0.f;
  for (int c = 0; c < CGRP; ++c) { float v = conv(c); sum += v; sq += v * v; }
  float mean = sum * (1.f / CGRP);
  float var  = sq * (1.f / CGRP) - mean * mean;
  float rstd = rsqrtf(var + 1e-5f);

  float oy = 0.f, ox = 0.f;
  for (int c = 0; c < CGRP; ++c) {
    float v  = conv(c);
    float tn = (v - mean) * rstd * ln_g[c] + ln_b[c];
    float ge = 0.5f * tn * (1.f + erff(tn * 0.7071067811865475f));   // exact GELU
    oy += pw_w[c] * ge;          // channel 0 = y
    ox += pw_w[CGRP + c] * ge;   // channel 1 = x
  }
  oy = tanhf(oy) * (2.f / 15.f);  // ORF * 1/(Hs-1)
  ox = tanhf(ox) * (2.f / 15.f);
  float ry = ((float)hs + 0.5f) * (2.f / 15.f) - 1.f;   // centered norm grid
  float rx = ((float)ws + 0.5f) * (2.f / 15.f) - 1.f;
  float py = oy + ry, px = ox + rx;
  size_t idx = ((size_t)(b * NGRP + g) * NS + t) * 2;
  posw[idx] = py;     posw[idx + 1] = px;
  out_pos[idx] = py;  out_pos[idx + 1] = px;
  out_ref[idx] = ry;  out_ref[idx + 1] = rx;
}

// ---------------------------------------------------------------------------
// Bilinear sample of x at pos -> xs_h[b][s][c]  (f16, sample-major)
// ---------------------------------------------------------------------------
__global__ __launch_bounds__(256)
void k_sample(const float* __restrict__ x, const float* __restrict__ posw,
              _Float16* __restrict__ xs_h) {
  int c = blockIdx.x * 256 + threadIdx.x;
  int s = blockIdx.y;
  int b = blockIdx.z;
  int g = c >> 7;
  size_t pi = ((size_t)(b * NGRP + g) * NS + s) * 2;
  float py = posw[pi], px = posw[pi + 1];
  float gx = (px + 1.f) * 0.5f * (WW - 1);
  float gy = (py + 1.f) * 0.5f * (HH - 1);
  float x0f = floorf(gx), y0f = floorf(gy);
  float fx = gx - x0f, fy = gy - y0f;
  int x0 = (int)x0f, y0 = (int)y0f;
  const float* img = x + ((size_t)b * NCH + c) * HWPIX;
  float w00 = (1.f - fx) * (1.f - fy), w10 = fx * (1.f - fy);
  float w01 = (1.f - fx) * fy,         w11 = fx * fy;
  float acc = 0.f;
  if (x0 >= 0     && x0 < WW     && y0 >= 0     && y0 < HH)     acc += w00 * img[y0 * WW + x0];
  if (x0 + 1 >= 0 && x0 + 1 < WW && y0 >= 0     && y0 < HH)     acc += w10 * img[y0 * WW + x0 + 1];
  if (x0 >= 0     && x0 < WW     && y0 + 1 >= 0 && y0 + 1 < HH) acc += w01 * img[(y0 + 1) * WW + x0];
  if (x0 + 1 >= 0 && x0 + 1 < WW && y0 + 1 >= 0 && y0 + 1 < HH) acc += w11 * img[(y0 + 1) * WW + x0 + 1];
  xs_h[((size_t)b * NS + s) * NCH + c] = (_Float16)acc;
}

// ---------------------------------------------------------------------------
// k_h[b][s][o] = sum_c xs_h[b][s][c] * Wk[o][c] + bk[o]   (f16 out)
// ---------------------------------------------------------------------------
__global__ __launch_bounds__(32)
void k_gemm_kproj(const _Float16* __restrict__ xs_h, const _Float16* __restrict__ wk,
                  const float* __restrict__ bk, _Float16* __restrict__ k_h) {
  int lane = threadIdx.x, col = lane & 15, hi = lane >> 4;
  int nb = blockIdx.x * 16;   // o tile (N)
  int sb = blockIdx.y * 16;   // sample tile (M)
  int b  = blockIdx.z;
  const _Float16* arow = xs_h + ((size_t)b * NS + sb + col) * NCH;
  const _Float16* brow = wk + (size_t)(nb + col) * NCH;
  v8f acc = {};
  for (int kb = 0; kb < NCH; kb += 32)
    acc = wmma_f16(ldA(arow + kb, hi), ldB(brow + kb, hi), acc);
  float bias = bk[nb + col];
  _Float16* dst = k_h + (size_t)b * NS * NCH;
#pragma unroll
  for (int r = 0; r < 8; ++r)
    dst[(size_t)(sb + r + 8 * hi) * NCH + nb + col] = (_Float16)(acc[r] + bias);
}

// ---------------------------------------------------------------------------
// v_h[b][o][s] = sum_c Wv[o][c] * xs_h[b][s][c] + bv[o]   (f16, channel-major)
// ---------------------------------------------------------------------------
__global__ __launch_bounds__(32)
void k_gemm_vproj(const _Float16* __restrict__ xs_h, const _Float16* __restrict__ wv,
                  const float* __restrict__ bv, _Float16* __restrict__ v_h) {
  int lane = threadIdx.x, col = lane & 15, hi = lane >> 4;
  int sn = blockIdx.x * 16;   // s tile (N)
  int ob = blockIdx.y * 16;   // o tile (M)
  int b  = blockIdx.z;
  const _Float16* arow = wv + (size_t)(ob + col) * NCH;
  const _Float16* brow = xs_h + ((size_t)b * NS + sn + col) * NCH;
  v8f acc = {};
  for (int kb = 0; kb < NCH; kb += 32)
    acc = wmma_f16(ldA(arow + kb, hi), ldB(brow + kb, hi), acc);
  _Float16* dst = v_h + (size_t)b * NCH * NS;
#pragma unroll
  for (int r = 0; r < 8; ++r) {
    int o = ob + r + 8 * hi;
    dst[(size_t)o * NS + sn + col] = (_Float16)(acc[r] + bv[o]);
  }
}

// ---------------------------------------------------------------------------
// RPE bias: bilinear (align_corners, zero-pad) lookup of 63x63 table in LDS.
// ---------------------------------------------------------------------------
__device__ __forceinline__ float rpe_bias(const float* s_tab, const float* s_pos,
                                          float qy, float qx, int sidx) {
  float py = s_pos[sidx * 2], px = s_pos[sidx * 2 + 1];
  float dy = (qy - py) * 0.5f, dx = (qx - px) * 0.5f;
  float gx = (dx + 1.f) * 31.f;   // (dx+1)*0.5*(63-1)
  float gy = (dy + 1.f) * 31.f;
  float x0f = floorf(gx), y0f = floorf(gy);
  float fx = gx - x0f, fy = gy - y0f;
  int x0 = (int)x0f, y0 = (int)y0f;
  float w00 = (1.f - fx) * (1.f - fy), w10 = fx * (1.f - fy);
  float w01 = (1.f - fx) * fy,         w11 = fx * fy;
  float acc = 0.f;
  if (x0 >= 0     && x0 < TABW     && y0 >= 0     && y0 < TABW)     acc += w00 * s_tab[y0 * TABW + x0];
  if (x0 + 1 >= 0 && x0 + 1 < TABW && y0 >= 0     && y0 < TABW)     acc += w10 * s_tab[y0 * TABW + x0 + 1];
  if (x0 >= 0     && x0 < TABW     && y0 + 1 >= 0 && y0 + 1 < TABW) acc += w01 * s_tab[(y0 + 1) * TABW + x0];
  if (x0 + 1 >= 0 && x0 + 1 < TABW && y0 + 1 >= 0 && y0 + 1 < TABW) acc += w11 * s_tab[(y0 + 1) * TABW + x0 + 1];
  return acc;
}

// ---------------------------------------------------------------------------
// Fused attention: per (b, head, 16-query tile). Flash-style online softmax
// over ns=256 in chunks of 32 (one WMMA-K). out_h[b][m][h*64+d] (f16).
// ---------------------------------------------------------------------------
__global__ __launch_bounds__(32)
void k_attn(const _Float16* __restrict__ q_h, const _Float16* __restrict__ k_h,
            const _Float16* __restrict__ v_h, const float* __restrict__ rpe,
            const float* __restrict__ posw, _Float16* __restrict__ out_h) {
  __shared__ float    s_tab[TABW * TABW];
  __shared__ float    s_pos[NS * 2];
  __shared__ _Float16 s_p[16 * 32];

  int lane = threadIdx.x, col = lane & 15, hi = lane >> 4;
  int mb = blockIdx.x * 16;
  int h  = blockIdx.y;
  int b  = blockIdx.z;
  int g  = h >> 1;   // GH = 2 heads per group

  for (int i = lane; i < TABW * TABW; i += 32) s_tab[i] = rpe[(size_t)h * TABW * TABW + i];
  {
    const float* pp = posw + (size_t)(b * NGRP + g) * NS * 2;
    for (int i = lane; i < NS * 2; i += 32) s_pos[i] = pp[i];
  }
  __syncthreads();

  // Q tile in A layout; K-dim = 64 -> two v16h
  const _Float16* qrow = q_h + ((size_t)b * HWPIX + mb + col) * NCH + h * DHD;
  v16h aq0 = ldA(qrow, hi), aq1 = ldA(qrow + 32, hi);

  float mrow[8], lrow[8], qy[8], qx[8];
  v8f acc[4];
  v8f vzero = {};
#pragma unroll
  for (int r = 0; r < 8; ++r) {
    mrow[r] = -__builtin_inff(); lrow[r] = 0.f;
    int m = mb + r + 8 * hi;
    qy[r] = (float)(m >> 5) * (2.f / 31.f) - 1.f;   // norm_grid(32, uncentered)
    qx[r] = (float)(m & 31) * (2.f / 31.f) - 1.f;
  }
#pragma unroll
  for (int dt = 0; dt < 4; ++dt) acc[dt] = vzero;

  const float scale = 0.125f;   // DH^-0.5
  const _Float16* kbase = k_h + (size_t)b * NS * NCH + h * DHD;
  const _Float16* vbase = v_h + (size_t)b * NCH * NS + (size_t)h * DHD * NS;

  for (int np = 0; np < NS; np += 32) {
    v8f s0 = vzero, s1 = vzero;
    {
      const _Float16* kr0 = kbase + (size_t)(np + col) * NCH;
      const _Float16* kr1 = kbase + (size_t)(np + 16 + col) * NCH;
      s0 = wmma_f16(aq0, ldB(kr0, hi), s0);
      s0 = wmma_f16(aq1, ldB(kr0 + 32, hi), s0);
      s1 = wmma_f16(aq0, ldB(kr1, hi), s1);
      s1 = wmma_f16(aq1, ldB(kr1 + 32, hi), s1);
    }

    float alpha[8];
#pragma unroll
    for (int r = 0; r < 8; ++r) {
      float e0 = s0[r] * scale + rpe_bias(s_tab, s_pos, qy[r], qx[r], np + col);
      float e1 = s1[r] * scale + rpe_bias(s_tab, s_pos, qy[r], qx[r], np + 16 + col);
      float mnew = fmaxf(mrow[r], hmax16(fmaxf(e0, e1)));
      float a  = __expf(mrow[r] - mnew);
      float p0 = __expf(e0 - mnew), p1 = __expf(e1 - mnew);
      lrow[r] = lrow[r] * a + hsum16(p0 + p1);
      mrow[r] = mnew; alpha[r] = a;
      s_p[(r + 8 * hi) * 32 + col]      = (_Float16)p0;
      s_p[(r + 8 * hi) * 32 + 16 + col] = (_Float16)p1;
    }
#pragma unroll
    for (int dt = 0; dt < 4; ++dt)
#pragma unroll
      for (int r = 0; r < 8; ++r) acc[dt][r] *= alpha[r];

    __syncthreads();
    // P back out of LDS in A layout (row = col, K = local sample index)
    v16h pa = ldA(&s_p[col * 32], hi);     // ds_load_b128 x2
    // PV: B[k=n'][n=d] = v_h[b][h*64 + dt*16 + d][np + n']  (contiguous in k)
#pragma unroll
    for (int dt = 0; dt < 4; ++dt) {
      const _Float16* vr = vbase + (size_t)(dt * 16 + col) * NS + np;
      acc[dt] = wmma_f16(pa, ldB(vr, hi), acc[dt]);
    }
    __syncthreads();
  }

  _Float16* orow = out_h + (size_t)b * HWPIX * NCH + h * DHD;
#pragma unroll
  for (int dt = 0; dt < 4; ++dt)
#pragma unroll
    for (int r = 0; r < 8; ++r) {
      int m = mb + r + 8 * hi;
      orow[(size_t)m * NCH + dt * 16 + col] = (_Float16)(acc[dt][r] / lrow[r]);
    }
}

// ---------------------------------------------------------------------------
// y[b][o][p] = sum_c Wo[o][c] * out_h[b][p][c] + bo[o]   (f32 final output)
// ---------------------------------------------------------------------------
__global__ __launch_bounds__(32)
void k_gemm_out(const _Float16* __restrict__ out_h, const _Float16* __restrict__ wo,
                const float* __restrict__ bo, float* __restrict__ y) {
  int lane = threadIdx.x, col = lane & 15, hi = lane >> 4;
  int pb = blockIdx.x * 16;   // pixel tile (N)
  int ob = blockIdx.y * 16;   // out-channel tile (M)
  int b  = blockIdx.z;
  const _Float16* arow = wo + (size_t)(ob + col) * NCH;
  const _Float16* brow = out_h + ((size_t)b * HWPIX + pb + col) * NCH;
  v8f acc = {};
  for (int kb = 0; kb < NCH; kb += 32)
    acc = wmma_f16(ldA(arow + kb, hi), ldB(brow + kb, hi), acc);
  float* dst = y + (size_t)b * NCH * HWPIX;
#pragma unroll
  for (int r = 0; r < 8; ++r) {
    int o = ob + r + 8 * hi;
    dst[(size_t)o * HWPIX + pb + col] = acc[r] + bo[o];
  }
}

// ---------------------------------------------------------------------------
extern "C" void kernel_launch(void* const* d_in, const int* in_sizes, int n_in,
                              void* d_out, int out_size, void* d_ws, size_t ws_size,
                              hipStream_t stream) {
  const float* x    = (const float*)d_in[0];
  const float* Wq   = (const float*)d_in[1];
  const float* bq   = (const float*)d_in[2];
  const float* Wk   = (const float*)d_in[3];
  const float* bk   = (const float*)d_in[4];
  const float* Wv   = (const float*)d_in[5];
  const float* bv   = (const float*)d_in[6];
  const float* Wo   = (const float*)d_in[7];
  const float* bo   = (const float*)d_in[8];
  const float* dw_w = (const float*)d_in[9];
  const float* dw_b = (const float*)d_in[10];
  const float* ln_g = (const float*)d_in[11];
  const float* ln_b = (const float*)d_in[12];
  const float* pw_w = (const float*)d_in[13];
  const float* rpe  = (const float*)d_in[14];

  char* w = (char*)d_ws;
  float* q_t = (float*)w;      w += (size_t)8388608 * 4;   // [16][1024][512] f32
  float* posw = (float*)w;     w += (size_t)32768 * 4;     // [16][4][256][2] f32
  _Float16* x_t  = (_Float16*)w; w += (size_t)8388608 * 2; // [16][1024][512] f16
  _Float16* q_h  = (_Float16*)w; w += (size_t)8388608 * 2; // [16][1024][512] f16
  _Float16* outh = (_Float16*)w; w += (size_t)8388608 * 2; // [16][1024][512] f16
  _Float16* xs_h = (_Float16*)w; w += (size_t)2097152 * 2; // [16][256][512]  f16
  _Float16* k_h  = (_Float16*)w; w += (size_t)2097152 * 2; // [16][256][512]  f16
  _Float16* v_h  = (_Float16*)w; w += (size_t)2097152 * 2; // [16][512][256]  f16
  _Float16* wq_h = (_Float16*)w; w += (size_t)262144 * 2;  // [512][512] f16
  _Float16* wk_h = (_Float16*)w; w += (size_t)262144 * 2;
  _Float16* wv_h = (_Float16*)w; w += (size_t)262144 * 2;
  _Float16* wo_h = (_Float16*)w; w += (size_t)262144 * 2;

  float* y       = (float*)d_out;      // [16][512][32][32]
  float* out_pos = y + 8388608;        // [16][4][16][16][2]
  float* out_ref = out_pos + 32768;    // [16][4][16][16][2]

  // One-shot operand staging (f16).
  k_cvt_f16<<<1024, 256, 0, stream>>>(Wq, wq_h, 262144);
  k_cvt_f16<<<1024, 256, 0, stream>>>(Wk, wk_h, 262144);
  k_cvt_f16<<<1024, 256, 0, stream>>>(Wv, wv_h, 262144);
  k_cvt_f16<<<1024, 256, 0, stream>>>(Wo, wo_h, 262144);
  k_xpose  <<<dim3(2, 1024, 16), 256, 0, stream>>>(x, x_t);

  k_gemm_q    <<<dim3(32, 64, 16), 32,  0, stream>>>(x_t, wq_h, bq, q_t, q_h);
  k_offset    <<<dim3(64),        256,  0, stream>>>(q_t, dw_w, dw_b, ln_g, ln_b, pw_w,
                                                     posw, out_pos, out_ref);
  k_sample    <<<dim3(2, 256, 16), 256, 0, stream>>>(x, posw, xs_h);
  k_gemm_kproj<<<dim3(32, 16, 16), 32,  0, stream>>>(xs_h, wk_h, bk, k_h);
  k_gemm_vproj<<<dim3(16, 32, 16), 32,  0, stream>>>(xs_h, wv_h, bv, v_h);
  k_attn      <<<dim3(64, 8, 16),  32,  0, stream>>>(q_h, k_h, v_h, rpe, posw, outh);
  k_gemm_out  <<<dim3(64, 32, 16), 32,  0, stream>>>(outh, wo_h, bo, y);
}
